// MatrixAttn_1314259993034
// MI455X (gfx1250) — compile-verified, matrix operand order checked by reading, not run on hardware
//
#include <hip/hip_runtime.h>
#include <hip/hip_bf16.h>

#define B_    16
#define N_    2048
#define M_    2048
#define DIN_  1024
#define DOUT_ 512

typedef __bf16 bf16;
typedef __attribute__((ext_vector_type(16))) __bf16 v16bf;
typedef __attribute__((ext_vector_type(8)))  __bf16 v8bf;
typedef __attribute__((ext_vector_type(8)))  float  v8f;
typedef __attribute__((ext_vector_type(4)))  unsigned u32x4;
typedef __attribute__((ext_vector_type(8)))  unsigned u32x8;

// ---- fragment helpers -------------------------------------------------------
// 16-bit A-matrix 16x32 layout (ISA 7.12.2): lanes 0-15 hold K = [0..7] in
// VGPR0-3 and K = [16..23] in VGPR4-7; lanes 16-31 hold K = [8..15], [24..31].
// So per lane: two contiguous 8-element (16-byte) chunks at k0+hi*8 and
// k0+hi*8+16. B columns use the mirrored layout (column index = lane&15).

__device__ __forceinline__ v8bf ld8bf(const bf16* p) {
  union { uint4 u; v8bf v; } c;
  c.u = *(const uint4*)p;          // 16B -> global_load_b128 / ds_load_b128
  return c.v;
}

__device__ __forceinline__ v8bf cvt8(const float* p) {
  union { uint4 u; float f[4]; } a, b;
  a.u = *(const uint4*)p;          // 2 x b128 loads
  b.u = *(const uint4*)(p + 4);
  v8bf r;
#pragma unroll
  for (int i = 0; i < 4; ++i) { r[i] = (bf16)a.f[i]; r[4 + i] = (bf16)b.f[i]; }
  return r;
}

__device__ __forceinline__ v16bf combine(v8bf lo, v8bf hif) {
  return __builtin_shufflevector(lo, hif, 0,1,2,3,4,5,6,7,8,9,10,11,12,13,14,15);
}

__device__ __forceinline__ v8f wmma_bf16(v16bf a, v16bf b, v8f c) {
  return __builtin_amdgcn_wmma_f32_16x16x32_bf16(
      /*neg_a=*/false, a, /*neg_b=*/false, b,
      /*c_mod=*/(short)0, c, /*reuse_a=*/false, /*reuse_b=*/false);
}

// ---- K1: emb f32 -> emb_bf16 (row major) + embT_bf16 (transposed) ----------
__global__ __launch_bounds__(256) void k_convert(const float* __restrict__ emb,
                                                 bf16* __restrict__ embbf,
                                                 bf16* __restrict__ embT) {
  __shared__ bf16 tile[64][72];            // padded to dodge bank conflicts
  const int b  = blockIdx.z;
  const int m0 = blockIdx.y * 64;
  const int e0 = blockIdx.x * 64;
  const int tid = threadIdx.x;
#pragma unroll
  for (int i = 0; i < 16; ++i) {
    const int idx = i * 256 + tid;
    const int m = idx >> 6, e = idx & 63;
    const float x = emb[((size_t)b * M_ + m0 + m) * DOUT_ + e0 + e];
    const bf16 h = (bf16)x;
    tile[m][e] = h;
    embbf[((size_t)b * M_ + m0 + m) * DOUT_ + e0 + e] = h;
  }
  __syncthreads();
#pragma unroll
  for (int i = 0; i < 16; ++i) {
    const int idx = i * 256 + tid;
    const int e = idx >> 6, m = idx & 63;
    embT[((size_t)b * DOUT_ + e0 + e) * M_ + m0 + m] = tile[m][e];
  }
}

// ---- K2: decsmall = dec @ W^T + b  (bf16 out to workspace) ------------------
// One wave: 16 rows x 64 cols (4 WMMA acc tiles). WG = 8 waves = 128 rows.
__global__ __launch_bounds__(256) void k_linear(const float* __restrict__ dec,
                                                const float* __restrict__ W,
                                                const float* __restrict__ bias,
                                                bf16* __restrict__ dsm) {
  const int lane = threadIdx.x & 31, wave = threadIdx.x >> 5;
  const int r = lane & 15, hi = lane >> 4;
  const int gn = blockIdx.y * 128 + wave * 16;   // flattened b*N + n
  const int e0 = blockIdx.x * 64;

  const float* arow = dec + (size_t)(gn + r) * DIN_;
  const float* brow[4];
#pragma unroll
  for (int t = 0; t < 4; ++t) brow[t] = W + (size_t)(e0 + t * 16 + r) * DIN_;

  v8f acc[4] = {};
  for (int k0 = 0; k0 < DIN_; k0 += 32) {
    __builtin_prefetch(arow + k0 + 64, 0, 0);    // global_prefetch_b8
    v16bf a = combine(cvt8(arow + k0 + hi * 8), cvt8(arow + k0 + hi * 8 + 16));
#pragma unroll
    for (int t = 0; t < 4; ++t) {
      v16bf bb = combine(cvt8(brow[t] + k0 + hi * 8),
                         cvt8(brow[t] + k0 + hi * 8 + 16));
      acc[t] = wmma_bf16(a, bb, acc[t]);
    }
  }
#pragma unroll
  for (int t = 0; t < 4; ++t) {
    const int e = e0 + t * 16 + r;
    const float bt = bias[e];
#pragma unroll
    for (int v = 0; v < 8; ++v)   // D: row = v + 8*hi, col = lane&15
      dsm[(size_t)(gn + v + 8 * hi) * DOUT_ + e] = (bf16)(acc[t][v] + bt);
  }
}

// ---- K3: scores -> masked softmax -> attn out + context GEMM ----------------
// WG = 16 N-rows x full M (2048). 8 waves, each owns a 256-wide M slab kept
// in 16 f32 WMMA accumulators. The decsmall A-tile (16x512 bf16 = 16KB) is
// DMA'd into LDS once per WG by the Tensor Data Mover; attn is staged bf16
// in a 64KB LDS tile for the phase-3 context GEMM.
__global__ __launch_bounds__(256) void k_attn(const bf16* __restrict__ dsm,
                                              const bf16* __restrict__ embbf,
                                              const bf16* __restrict__ embT,
                                              const long long* __restrict__ elen,
                                              float* __restrict__ attn,
                                              float* __restrict__ out) {
  __shared__ bf16 a_lds[16][DOUT_];                 // 16384 bytes (TDM target)
  __shared__ bf16 attn_lds[16][M_];                 // 65536 bytes
  float* red = (float*)&attn_lds[0][0];             // reused before bf16 stage

  const int lane = threadIdx.x & 31, wave = threadIdx.x >> 5;
  const int r = lane & 15, hi = lane >> 4;
  const int b  = blockIdx.x >> 7;                   // N_/16 == 128 tiles per b
  const int n0 = (blockIdx.x & 127) * 16;
  const int mbase = wave * 256;
  const size_t rowbase = (size_t)b * N_ + n0;

  // ---- TDM: stage decsmall[rowbase .. rowbase+15][0..511] bf16 into LDS ----
  if (wave == 0) {
    const unsigned long long gaddr =
        (unsigned long long)(const void*)(dsm + rowbase * DOUT_);
    const unsigned lds_off =
        (unsigned)(unsigned long long)(const void*)&a_lds[0][0];
    u32x4 g0; u32x8 g1; u32x4 g2; u32x4 g3;
    // D# group0: count=1 (valid), lds_addr, 57-bit global_addr, type=2
    g0[0] = 1u;
    g0[1] = lds_off;
    g0[2] = (unsigned)gaddr;
    g0[3] = (unsigned)((gaddr >> 32) & 0x01FFFFFFu) | (2u << 30);
    // D# group1: data_size=2B; tensor_dim0=512, tensor_dim1=16;
    //            tile_dim0=512, tile_dim1=16, tile_dim2=0; dim0_stride=512
    g1[0] = 0x00010000u;                 // wg_mask=0, data_size=1 (2 bytes)
    g1[1] = (unsigned)DOUT_ << 16;       // tensor_dim0[15:0] << 16
    g1[2] = 16u << 16;                   // tensor_dim1[15:0] << 16
    g1[3] = (unsigned)DOUT_ << 16;       // tile_dim0 << 16
    g1[4] = 16u;                         // tile_dim1=16, tile_dim2=0
    g1[5] = (unsigned)DOUT_;             // tensor_dim0_stride lo32
    g1[6] = 0u;
    g1[7] = 0u;
    // D# groups 2/3: benign (tile_dim2=0 => higher dims unused)
    g2[0] = 1u; g2[1] = 1u; g2[2] = 0u; g2[3] = 0u;
    g3[0] = 0u; g3[1] = 0u; g3[2] = 0u; g3[3] = 0u;
    asm volatile("tensor_load_to_lds %0, %1, %2, %3"
                 :: "s"(g0), "s"(g1), "s"(g2), "s"(g3) : "memory");
    __builtin_amdgcn_s_wait_tensorcnt(0);
  }
  __syncthreads();

  // ---- phase 1: scores = decsmall(16xDOUT) @ emb^T, slab of 256 m-cols ----
  v8f acc[16] = {};
  {
    const bf16* arow = &a_lds[r][0];                // A fed from LDS
    for (int k0 = 0; k0 < DOUT_; k0 += 32) {
      v16bf a = combine(ld8bf(arow + k0 + hi * 8),
                        ld8bf(arow + k0 + hi * 8 + 16));
#pragma unroll
      for (int t = 0; t < 16; ++t) {
        const bf16* brow = embbf + ((size_t)b * M_ + mbase + t * 16 + r) * DOUT_;
        v16bf bb = combine(ld8bf(brow + k0 + hi * 8),
                           ld8bf(brow + k0 + hi * 8 + 16));
        acc[t] = wmma_bf16(a, bb, acc[t]);
      }
    }
  }

  // ---- mask: scores[:, m >= elen[b]] = -inf -------------------------------
  const long long len = elen[b];
#pragma unroll
  for (int t = 0; t < 16; ++t)
    if ((long long)(mbase + t * 16 + r) >= len) {
#pragma unroll
      for (int v = 0; v < 8; ++v) acc[t][v] = -__builtin_inff();
    }

  // ---- phase 2: softmax over M (half-wave shuffles + LDS cross-wave) ------
  float rmax[8], rscale[8];
#pragma unroll
  for (int v = 0; v < 8; ++v) {
    float m = acc[0][v];
#pragma unroll
    for (int t = 1; t < 16; ++t) m = fmaxf(m, acc[t][v]);
    for (int s = 1; s < 16; s <<= 1) m = fmaxf(m, __shfl_xor(m, s, 32));
    if (r == 0) red[(v + 8 * hi) * 8 + wave] = m;   // per-wave partial max
  }
  __syncthreads();
#pragma unroll
  for (int v = 0; v < 8; ++v) {
    float m = red[(v + 8 * hi) * 8];
#pragma unroll
    for (int w = 1; w < 8; ++w) m = fmaxf(m, red[(v + 8 * hi) * 8 + w]);
    rmax[v] = m;
  }

  float psum[8];
#pragma unroll
  for (int v = 0; v < 8; ++v) psum[v] = 0.0f;
#pragma unroll
  for (int t = 0; t < 16; ++t) {
#pragma unroll
    for (int v = 0; v < 8; ++v) {
      const float x = __expf(acc[t][v] - rmax[v]);  // v_exp_f32
      acc[t][v] = x;
      psum[v] += x;
    }
  }
#pragma unroll
  for (int v = 0; v < 8; ++v) {
    float s = psum[v];
    for (int i = 1; i < 16; i <<= 1) s += __shfl_xor(s, i, 32);
    if (r == 0) red[128 + (v + 8 * hi) * 8 + wave] = s;
  }
  __syncthreads();
#pragma unroll
  for (int v = 0; v < 8; ++v) {
    float s = 0.0f;
#pragma unroll
    for (int w = 0; w < 8; ++w) s += red[128 + (v + 8 * hi) * 8 + w];
    rscale[v] = 1.0f / s;
  }
  __syncthreads();   // all red[] reads done before attn_lds overwrite

  // ---- write attn (f32, to d_out) and stage bf16 attn tile in LDS ---------
#pragma unroll
  for (int t = 0; t < 16; ++t) {
    const int m = mbase + t * 16 + r;
#pragma unroll
    for (int v = 0; v < 8; ++v) {
      const int row = v + 8 * hi;
      const float a = acc[t][v] * rscale[v];
      attn[(rowbase + row) * M_ + m] = a;
      attn_lds[row][m] = (bf16)a;
    }
  }
  __syncthreads();

  // ---- phase 3: out(16xDOUT) = attn(16xM) @ emb(MxDOUT), via embT ---------
  v8f oacc[4] = {};
  const int e0 = wave * 64;
  {
    const bf16* arow = &attn_lds[r][0];             // A from LDS (ds_load_b128)
    for (int k0 = 0; k0 < M_; k0 += 32) {
      v16bf a = combine(ld8bf(arow + k0 + hi * 8),
                        ld8bf(arow + k0 + hi * 8 + 16));
#pragma unroll
      for (int t = 0; t < 4; ++t) {
        const bf16* brow = embT + ((size_t)b * DOUT_ + e0 + t * 16 + r) * M_;
        v16bf bb = combine(ld8bf(brow + k0 + hi * 8),
                           ld8bf(brow + k0 + hi * 8 + 16));
        oacc[t] = wmma_bf16(a, bb, oacc[t]);
      }
    }
  }
#pragma unroll
  for (int t = 0; t < 4; ++t) {
#pragma unroll
    for (int v = 0; v < 8; ++v)
      out[(rowbase + v + 8 * hi) * DOUT_ + e0 + t * 16 + r] = oacc[t][v];
  }
}

// ---- launcher ---------------------------------------------------------------
extern "C" void kernel_launch(void* const* d_in, const int* in_sizes, int n_in,
                              void* d_out, int out_size, void* d_ws, size_t ws_size,
                              hipStream_t stream) {
  const float*     dec  = (const float*)d_in[0];      // [B,N,DIN]
  const float*     emb  = (const float*)d_in[1];      // [B,M,DOUT]
  const long long* elen = (const long long*)d_in[2];  // [B] int64
  const float*     W    = (const float*)d_in[3];      // [DOUT,DIN]
  const float*     bias = (const float*)d_in[4];      // [DOUT]

  float* out  = (float*)d_out;                        // [B,N,DOUT]
  float* attn = out + (size_t)B_ * N_ * DOUT_;        // [B,N,M]

  // workspace: decsmall_bf16 | emb_bf16 | embT_bf16  (~100 MB total)
  bf16* dsm   = (bf16*)d_ws;
  bf16* embbf = dsm   + (size_t)B_ * N_ * DOUT_;
  bf16* embT  = embbf + (size_t)B_ * M_ * DOUT_;

  k_convert<<<dim3(DOUT_ / 64, M_ / 64, B_), 256, 0, stream>>>(emb, embbf, embT);
  k_linear<<<dim3(DOUT_ / 64, (B_ * N_) / 128), 256, 0, stream>>>(dec, W, bias, dsm);
  k_attn<<<dim3((B_ * N_) / 16), 256, 0, stream>>>(dsm, embbf, embT, elen, attn, out);
}
